// MoDBlock_65687229825451
// MI455X (gfx1250) — compile-verified, matrix operand order checked by reading, not run on hardware
//
#include <hip/hip_runtime.h>

// ---------------------------------------------------------------------------
// MoD transformer block for MI455X (gfx1250, wave32, WMMA bf16 + TDM)
// ---------------------------------------------------------------------------
#define BATCH 4
#define TTOK  4096
#define DMODEL 1024
#define NHEAD 16
#define DHEAD 64
#define KTOK  2048           // TTOK * CAP
#define MROWS (BATCH * KTOK) // 8192 routed rows

typedef __attribute__((ext_vector_type(16))) __bf16 v16bf;
typedef __attribute__((ext_vector_type(8)))  float  v8f;

union F8 { v8f v; float f[8]; };

#if defined(__has_builtin)
#if __has_builtin(__builtin_amdgcn_tensor_load_to_lds)
#define HAVE_TDM 1
#endif
#if __has_builtin(__builtin_amdgcn_s_wait_tensorcnt)
#define HAVE_TCNT 1
#endif
#endif

__device__ __forceinline__ __bf16 to_bf16(float x) {
    unsigned u = __float_as_uint(x);
    unsigned r = (u + 0x7FFFu + ((u >> 16) & 1u)) >> 16;   // RNE
    unsigned short hs = (unsigned short)r;
    __bf16 out;
    __builtin_memcpy(&out, &hs, 2);
    return out;
}

// Load a 16x32 bf16 fragment (A or B role) from LDS laid out [row][k], stride ld.
// ISA 7.12.2: lanes 0-15 -> rows 0-15 with K = {0..7,16..23}+ks, lanes 16-31
// same rows with K = {8..15,24..31}+ks; each dword holds a (k,k+1) bf16 pair.
__device__ __forceinline__ v16bf frag16(const __bf16* base, int row0, int ks,
                                        int lane, int ld) {
    union { v16bf v; unsigned u[8]; } f;
    const int g = lane >> 4;
    const __bf16* p = base + (size_t)(row0 + (lane & 15)) * ld + ks + g * 8;
#pragma unroll
    for (int i = 0; i < 8; ++i) {
        int k = ((i & 3) << 1) + ((i >> 2) << 4);
        f.u[i] = *(const unsigned*)(p + k);
    }
    return f.v;
}

// ---------------------------------------------------------------------------
// Tensor Data Mover: DMA one [128 rows x 64 cols] bf16 tile (row stride =
// stride_elems) from global into LDS with a 4-dword pad after each 32-dword
// row (LDS row pitch 72 bf16 = LDT). Issued by one wave; TENSORcnt tracked.
// ---------------------------------------------------------------------------
#if HAVE_TDM
typedef __attribute__((ext_vector_type(4))) unsigned tdm_u4;
typedef __attribute__((ext_vector_type(8))) int      tdm_i8;
typedef __attribute__((ext_vector_type(4))) int      tdm_i4;

__device__ __forceinline__ void tdm_load_tile(const __bf16* gsrc, unsigned lds_off,
                                              unsigned stride_elems) {
    const unsigned long long ga = (unsigned long long)(size_t)gsrc;
    tdm_u4 g0;
    g0[0] = 1u;                                  // D# count = 1 (valid user desc)
    g0[1] = lds_off;                             // lds_addr (bytes)
    g0[2] = (unsigned)ga;                        // global_addr[31:0]
    g0[3] = (unsigned)(ga >> 32) | (2u << 30);   // global_addr[56:32] | type=2
    const unsigned td0 = stride_elems;           // tensor_dim0 = row length (K)
    const unsigned td1 = 1u << 20;               // tensor_dim1: large, no OOB
    tdm_i8 g1;
    g1[0] = (int)((1u << 16) |                   // data_size = 2 bytes
                  (1u << 20) |                   // pad_enable
                  (4u << 22) |                   // pad_interval = 32 dwords
                  (3u << 25));                   // pad_amount   = 4 dwords
    g1[1] = (int)((td0 & 0xFFFFu) << 16);                 // tensor_dim0[15:0]
    g1[2] = (int)((td0 >> 16) | ((td1 & 0xFFFFu) << 16)); // d0[31:16]|d1[15:0]
    g1[3] = (int)((td1 >> 16) | (64u << 16));             // d1[31:16]|tile_dim0=64
    g1[4] = 128;                                          // tile_dim1=128, tile_dim2=0
    g1[5] = (int)stride_elems;                            // dim0_stride[31:0]
    g1[6] = 0;                                            // dim0_stride[47:32]
    g1[7] = 0;
    const tdm_i4 z4 = {0, 0, 0, 0};
#if __clang_major__ >= 23
    const tdm_i8 z8 = {0, 0, 0, 0, 0, 0, 0, 0};
    __builtin_amdgcn_tensor_load_to_lds(g0, g1, z4, z4, z8, 0);
#else
    __builtin_amdgcn_tensor_load_to_lds(g0, g1, z4, z4, 0);
#endif
}
#endif

__device__ __forceinline__ void tdm_wait0() {
#if HAVE_TCNT
    __builtin_amdgcn_s_wait_tensorcnt(0);
#else
    asm volatile("s_wait_tensorcnt 0x0" ::: "memory");
#endif
}

// ---------------------------------------------------------------------------
// 1) Router: logits[b,t] = x[b,t,:] . w_router ; accumulate sum(logits^2)
// ---------------------------------------------------------------------------
__global__ __launch_bounds__(256)
void router_kernel(const float* __restrict__ x, const float* __restrict__ wr,
                   float* __restrict__ logits, float* __restrict__ auxacc) {
    const int lane = threadIdx.x & 31, wid = threadIdx.x >> 5;
    const int token = blockIdx.x * 8 + wid;
    const float* xr = x + (size_t)token * DMODEL;
    float s = 0.f;
#pragma unroll 4
    for (int i = lane; i < DMODEL; i += 32) s += xr[i] * wr[i];
#pragma unroll
    for (int m = 16; m >= 1; m >>= 1) s += __shfl_xor(s, m, 32);
    __shared__ float part[8];
    if (lane == 0) { logits[token] = s; part[wid] = s * s; }
    __syncthreads();
    if (threadIdx.x == 0) {
        float a = 0.f;
#pragma unroll
        for (int j = 0; j < 8; ++j) a += part[j];
        atomicAdd(auxacc, a);
    }
}

__global__ void aux_kernel(const float* __restrict__ acc, float* __restrict__ out) {
    out[0] = 0.01f * acc[0] * (1.0f / (BATCH * TTOK));
}

// ---------------------------------------------------------------------------
// 2) Per-batch top-k (k = KTOK): radix-select threshold + ordered compaction.
// ---------------------------------------------------------------------------
__device__ __forceinline__ unsigned fkey(float x) {
    unsigned u = __float_as_uint(x);
    return (u & 0x80000000u) ? ~u : (u | 0x80000000u);
}

__global__ __launch_bounds__(256)
void topk_kernel(const float* __restrict__ logits, int* __restrict__ idx) {
    __shared__ int hist[256];
    __shared__ unsigned s_prefix, s_mask;
    __shared__ int s_need, s_base, s_ties;
    __shared__ int wTotE[8], wTotK[8];
    const int b = blockIdx.x, tid = threadIdx.x;
    const int lane = tid & 31, w = tid >> 5;
    const float* lg = logits + (size_t)b * TTOK;

    if (tid == 0) { s_prefix = 0u; s_mask = 0u; s_need = KTOK; }
    __syncthreads();
    for (int byte = 3; byte >= 0; --byte) {
        hist[tid] = 0;
        __syncthreads();
        unsigned pref = s_prefix, msk = s_mask;
        for (int i = tid; i < TTOK; i += 256) {
            unsigned key = fkey(lg[i]);
            if ((key & msk) == pref) atomicAdd(&hist[(key >> (byte * 8)) & 255], 1);
        }
        __syncthreads();
        if (tid == 0) {
            int cum = 0, need = s_need, chosen = 0;
            for (int v = 255; v >= 0; --v) {
                if (cum + hist[v] >= need) { chosen = v; break; }
                cum += hist[v];
            }
            s_need = need - cum;
            s_prefix = pref | ((unsigned)chosen << (byte * 8));
            s_mask = msk | (0xFFu << (byte * 8));
        }
        __syncthreads();
    }
    const unsigned thr = s_prefix;
    const int need = s_need;
    if (tid == 0) { s_base = 0; s_ties = 0; }
    __syncthreads();

    for (int start = 0; start < TTOK; start += 256) {
        const int i = start + tid;
        const unsigned key = fkey(lg[i]);
        const int isE = (key == thr);
        const int isG = (key > thr);
        unsigned mE = (unsigned)__ballot(isE);
        int exE = __popc(mE & ((1u << lane) - 1u));
        if (lane == 0) wTotE[w] = __popc(mE);
        __syncthreads();
        int baseE = 0;
        for (int j = 0; j < w; ++j) baseE += wTotE[j];
        const int tieRank = s_ties + baseE + exE;
        const int keep = isG || (isE && tieRank < need);
        unsigned mK = (unsigned)__ballot(keep);
        int exK = __popc(mK & ((1u << lane) - 1u));
        if (lane == 0) wTotK[w] = __popc(mK);
        __syncthreads();
        int baseK = 0;
        for (int j = 0; j < w; ++j) baseK += wTotK[j];
        const int pos = s_base + baseK + exK;
        if (keep) idx[(size_t)b * KTOK + pos] = i;
        __syncthreads();
        if (tid == 0) {
            int tE = 0, tK = 0;
            for (int j = 0; j < 8; ++j) { tE += wTotE[j]; tK += wTotK[j]; }
            s_ties += tE; s_base += tK;
        }
        __syncthreads();
    }
}

// ---------------------------------------------------------------------------
// 3) LayerNorm (optionally gathering rows through idx), bf16 output.
// ---------------------------------------------------------------------------
__device__ __forceinline__ float block_sum256(float v, volatile float* sb) {
#pragma unroll
    for (int m = 16; m >= 1; m >>= 1) v += __shfl_xor(v, m, 32);
    if ((threadIdx.x & 31) == 0) sb[threadIdx.x >> 5] = v;
    __syncthreads();
    float t = 0.f;
#pragma unroll
    for (int j = 0; j < 8; ++j) t += sb[j];
    return t;
}

__global__ __launch_bounds__(256)
void ln_kernel(const float* __restrict__ X, const int* __restrict__ idx,
               const float* __restrict__ gamma, const float* __restrict__ beta,
               float* __restrict__ xs_out, __bf16* __restrict__ h_out) {
    __shared__ float sb1[8], sb2[8];
    const int row = blockIdx.x;                 // 0 .. MROWS-1
    const float* xr;
    if (idx) {
        const int b = row / KTOK;
        const int t = idx[row];
        xr = X + ((size_t)b * TTOK + t) * DMODEL;
    } else {
        xr = X + (size_t)row * DMODEL;
    }
    float v[4], s = 0.f, q = 0.f;
#pragma unroll
    for (int j = 0; j < 4; ++j) {
        v[j] = xr[threadIdx.x + j * 256];
        s += v[j]; q += v[j] * v[j];
    }
    const float sum  = block_sum256(s, sb1);
    const float sums = block_sum256(q, sb2);
    const float mu = sum * (1.0f / DMODEL);
    const float var = sums * (1.0f / DMODEL) - mu * mu;
    const float rs = rsqrtf(var + 1e-5f);
#pragma unroll
    for (int j = 0; j < 4; ++j) {
        const int i = threadIdx.x + j * 256;
        if (xs_out) xs_out[(size_t)row * DMODEL + i] = v[j];
        h_out[(size_t)row * DMODEL + i] =
            to_bf16((v[j] - mu) * rs * gamma[i] + beta[i]);
    }
}

// ---------------------------------------------------------------------------
// 4) Tiled bf16 WMMA GEMM: C[M,N] = A[M,K] * W[N,K]^T + bias (+ resid)
//    Block = 256 threads (8 waves). Tile 128x128, BK=64. Wave tile 32x64.
//    A/B tiles staged to LDS by the Tensor Data Mover when available.
//    Requires M%128==0, N%128==0, K%64==0 (true for all call sites).
// ---------------------------------------------------------------------------
#define BK 64
#define LDT (BK + 8)

__global__ __launch_bounds__(256)
void gemm_bf16_kernel(const __bf16* __restrict__ A, const __bf16* __restrict__ W,
                      const float* __restrict__ bias, const float* __restrict__ resid,
                      float* __restrict__ C, int M, int N, int K) {
    __shared__ __align__(16) __bf16 As[128 * LDT];
    __shared__ __align__(16) __bf16 Bs[128 * LDT];
    const int tid = threadIdx.x, lane = tid & 31, wid = tid >> 5;
    const int m0 = blockIdx.y * 128, n0 = blockIdx.x * 128;
    const int wm = (wid & 3) * 32, wn = (wid >> 2) * 64;

    F8 acc[2][4];
#pragma unroll
    for (int a = 0; a < 2; ++a)
#pragma unroll
        for (int b = 0; b < 4; ++b)
#pragma unroll
            for (int j = 0; j < 8; ++j) acc[a][b].f[j] = 0.f;

    for (int k0 = 0; k0 < K; k0 += BK) {
        __syncthreads();                 // previous tile fully consumed
#if HAVE_TDM
        if (wid == 0) {
            tdm_load_tile(A + (size_t)m0 * K + k0, (unsigned)(size_t)As, (unsigned)K);
            tdm_load_tile(W + (size_t)n0 * K + k0, (unsigned)(size_t)Bs, (unsigned)K);
            tdm_wait0();                 // TENSORcnt == 0 -> tiles in LDS
        }
#else
#pragma unroll
        for (int c = 0; c < 4; ++c) {
            const int g = tid + c * 256;          // 0..1023
            const int row = g >> 3;
            const int col = (g & 7) * 8;
            *(uint4*)&As[row * LDT + col] =
                *(const uint4*)&A[(size_t)(m0 + row) * K + k0 + col];
            *(uint4*)&Bs[row * LDT + col] =
                *(const uint4*)&W[(size_t)(n0 + row) * K + k0 + col];
        }
#endif
        __syncthreads();
#pragma unroll
        for (int ks = 0; ks < BK; ks += 32) {
            v16bf a0 = frag16(As, wm,      ks, lane, LDT);
            v16bf a1 = frag16(As, wm + 16, ks, lane, LDT);
#pragma unroll
            for (int fn = 0; fn < 4; ++fn) {
                v16bf bb = frag16(Bs, wn + fn * 16, ks, lane, LDT);
                acc[0][fn].v = __builtin_amdgcn_wmma_f32_16x16x32_bf16(
                    false, a0, false, bb, (short)0, acc[0][fn].v, false, false);
                acc[1][fn].v = __builtin_amdgcn_wmma_f32_16x16x32_bf16(
                    false, a1, false, bb, (short)0, acc[1][fn].v, false, false);
            }
        }
    }
    const int g = lane >> 4;
    if (resid) {
#pragma unroll
        for (int fm = 0; fm < 2; ++fm)
#pragma unroll
            for (int fn = 0; fn < 4; ++fn) {
                const int rbase = m0 + wm + fm * 16 + g * 8;
                const int col   = n0 + wn + fn * 16 + (lane & 15);
                const float bb = bias ? bias[col] : 0.f;
#pragma unroll
                for (int j = 0; j < 8; ++j) {
                    const size_t off = (size_t)(rbase + j) * N + col;
                    C[off] = acc[fm][fn].f[j] + bb + resid[off];
                }
            }
    } else {
#pragma unroll
        for (int fm = 0; fm < 2; ++fm)
#pragma unroll
            for (int fn = 0; fn < 4; ++fn) {
                const int rbase = m0 + wm + fm * 16 + g * 8;
                const int col   = n0 + wn + fn * 16 + (lane & 15);
                const float bb = bias ? bias[col] : 0.f;
#pragma unroll
                for (int j = 0; j < 8; ++j) {
                    const size_t off = (size_t)(rbase + j) * N + col;
                    C[off] = acc[fm][fn].f[j] + bb;
                }
            }
    }
}

// ---------------------------------------------------------------------------
// 5) Flash attention: grid (KTOK/64, B*H), 128 threads = 4 waves, each wave
//    owns 16 query rows. S = QK^T and O += PV via WMMA bf16, online softmax.
// ---------------------------------------------------------------------------
#define ALD 72

__global__ __launch_bounds__(128)
void attn_kernel(const float* __restrict__ qkv, __bf16* __restrict__ obf) {
    __shared__ __align__(16) __bf16 Qs[64 * ALD];
    __shared__ __align__(16) __bf16 Ks[64 * ALD];
    __shared__ __align__(16) __bf16 Vt[64 * ALD];   // transposed: [dh][kv]
    __shared__ __align__(16) __bf16 Ps[64 * ALD];
    const int tid = threadIdx.x, lane = tid & 31, wid = tid >> 5;
    const int g = lane >> 4;
    const int b = blockIdx.y / NHEAD, h = blockIdx.y % NHEAD;
    const int q0 = blockIdx.x * 64;
    const size_t rstride = (size_t)3 * DMODEL;
    const float* base = qkv + (size_t)b * KTOK * rstride + (size_t)h * DHEAD;

    for (int i = tid; i < 64 * 64; i += 128) {
        const int r = i >> 6, c = i & 63;
        Qs[r * ALD + c] = to_bf16(base[(size_t)(q0 + r) * rstride + c] * 0.125f);
    }

    F8 o[4];
    float rowmax[8], rowsum[8];
#pragma unroll
    for (int f = 0; f < 4; ++f)
#pragma unroll
        for (int j = 0; j < 8; ++j) o[f].f[j] = 0.f;
#pragma unroll
    for (int j = 0; j < 8; ++j) { rowmax[j] = -1e30f; rowsum[j] = 0.f; }

    for (int kv0 = 0; kv0 < KTOK; kv0 += 64) {
        __syncthreads();
        for (int i = tid; i < 64 * 64; i += 128) {
            const int r = i >> 6, c = i & 63;
            const size_t ro = (size_t)(kv0 + r) * rstride;
            Ks[r * ALD + c] = to_bf16(base[ro + DMODEL + c]);
            Vt[c * ALD + r] = to_bf16(base[ro + 2 * DMODEL + c]);
        }
        __syncthreads();

        F8 s[4];
#pragma unroll
        for (int f = 0; f < 4; ++f)
#pragma unroll
            for (int j = 0; j < 8; ++j) s[f].f[j] = 0.f;
#pragma unroll
        for (int ks = 0; ks < 64; ks += 32) {
            v16bf aq = frag16(Qs, wid * 16, ks, lane, ALD);
#pragma unroll
            for (int f = 0; f < 4; ++f) {
                v16bf bk = frag16(Ks, f * 16, ks, lane, ALD);
                s[f].v = __builtin_amdgcn_wmma_f32_16x16x32_bf16(
                    false, aq, false, bk, (short)0, s[f].v, false, false);
            }
        }
        // online softmax update (rows g*8+j, columns across the 16 lanes of g)
#pragma unroll
        for (int j = 0; j < 8; ++j) {
            float m = fmaxf(fmaxf(s[0].f[j], s[1].f[j]),
                            fmaxf(s[2].f[j], s[3].f[j]));
#pragma unroll
            for (int msk = 8; msk >= 1; msk >>= 1)
                m = fmaxf(m, __shfl_xor(m, msk, 32));
            const float mn = fmaxf(rowmax[j], m);
            const float corr = __expf(rowmax[j] - mn);
            rowmax[j] = mn;
            rowsum[j] *= corr;
#pragma unroll
            for (int f = 0; f < 4; ++f) o[f].f[j] *= corr;
        }
#pragma unroll
        for (int f = 0; f < 4; ++f)
#pragma unroll
            for (int j = 0; j < 8; ++j) {
                const float p = __expf(s[f].f[j] - rowmax[j]);
                rowsum[j] += p;   // per-lane partial; reduced once at the end
                Ps[(wid * 16 + g * 8 + j) * ALD + f * 16 + (lane & 15)] = to_bf16(p);
            }
        // O += P * V  (each wave reads only its own 16 Ps rows -> no barrier)
#pragma unroll
        for (int ks = 0; ks < 64; ks += 32) {
            v16bf ap = frag16(Ps, wid * 16, ks, lane, ALD);
#pragma unroll
            for (int f = 0; f < 4; ++f) {
                v16bf bv = frag16(Vt, f * 16, ks, lane, ALD);
                o[f].v = __builtin_amdgcn_wmma_f32_16x16x32_bf16(
                    false, ap, false, bv, (short)0, o[f].v, false, false);
            }
        }
    }
#pragma unroll
    for (int j = 0; j < 8; ++j) {
        float t = rowsum[j];
#pragma unroll
        for (int msk = 8; msk >= 1; msk >>= 1) t += __shfl_xor(t, msk, 32);
        rowsum[j] = 1.0f / t;
    }
#pragma unroll
    for (int f = 0; f < 4; ++f)
#pragma unroll
        for (int j = 0; j < 8; ++j) {
            const int r = q0 + wid * 16 + g * 8 + j;
            const int c = h * DHEAD + f * 16 + (lane & 15);
            obf[((size_t)b * KTOK + r) * DMODEL + c] =
                to_bf16(o[f].f[j] * rowsum[j]);
        }
}

// ---------------------------------------------------------------------------
// 6) Elementwise helpers
// ---------------------------------------------------------------------------
__global__ __launch_bounds__(256)
void conv_bf16_kernel(const float* __restrict__ in, __bf16* __restrict__ out, int n) {
    int i = blockIdx.x * 256 + threadIdx.x;
    if (i < n) out[i] = to_bf16(in[i]);
}

__global__ __launch_bounds__(256)
void gelu_bf16_kernel(const float* __restrict__ in, __bf16* __restrict__ out, int n) {
    int i = blockIdx.x * 256 + threadIdx.x;
    if (i < n) {
        const float x = in[i];
        out[i] = to_bf16(0.5f * x * (1.0f + erff(x * 0.70710678118654752f)));
    }
}

__global__ __launch_bounds__(256)
void scatter_kernel(const float* __restrict__ xs, const int* __restrict__ idx,
                    float* __restrict__ out) {
    const int row = blockIdx.x;
    const int b = row / KTOK;
    const int t = idx[row];
    const float* src = xs + (size_t)row * DMODEL;
    float* dst = out + ((size_t)b * TTOK + t) * DMODEL;
#pragma unroll
    for (int j = 0; j < 4; ++j) dst[threadIdx.x + j * 256] = src[threadIdx.x + j * 256];
}

// ---------------------------------------------------------------------------
// Host launcher
// ---------------------------------------------------------------------------
extern "C" void kernel_launch(void* const* d_in, const int* in_sizes, int n_in,
                              void* d_out, int out_size, void* d_ws, size_t ws_size,
                              hipStream_t stream) {
    (void)in_sizes; (void)n_in; (void)out_size; (void)ws_size;
    const float* x        = (const float*)d_in[0];
    const float* w_router = (const float*)d_in[1];
    const float* in_w     = (const float*)d_in[2];
    const float* in_b     = (const float*)d_in[3];
    const float* out_w    = (const float*)d_in[4];
    const float* out_b    = (const float*)d_in[5];
    const float* w1       = (const float*)d_in[6];
    const float* b1       = (const float*)d_in[7];
    const float* w2       = (const float*)d_in[8];
    const float* b2       = (const float*)d_in[9];
    const float* ln1_g    = (const float*)d_in[10];
    const float* ln1_b    = (const float*)d_in[11];
    const float* ln2_g    = (const float*)d_in[12];
    const float* ln2_b    = (const float*)d_in[13];
    float* out = (float*)d_out;

    // workspace layout
    char* p = (char*)d_ws;
    auto alloc = [&](size_t bytes) {
        char* r = p; p += (bytes + 255) & ~(size_t)255; return (void*)r;
    };
    float*   logits  = (float*)  alloc((size_t)BATCH * TTOK * 4);
    float*   auxacc  = (float*)  alloc(256);
    int*     idx     = (int*)    alloc((size_t)BATCH * KTOK * 4);
    float*   xs      = (float*)  alloc((size_t)MROWS * DMODEL * 4);
    __bf16*  h1      = (__bf16*) alloc((size_t)MROWS * DMODEL * 2);
    __bf16*  wqkv    = (__bf16*) alloc((size_t)3 * DMODEL * DMODEL * 2);
    __bf16*  wout    = (__bf16*) alloc((size_t)DMODEL * DMODEL * 2);
    __bf16*  w1b     = (__bf16*) alloc((size_t)4 * DMODEL * DMODEL * 2);
    __bf16*  w2b     = (__bf16*) alloc((size_t)4 * DMODEL * DMODEL * 2);
    float*   qkv     = (float*)  alloc((size_t)MROWS * 3 * DMODEL * 4);
    __bf16*  obf     = (__bf16*) alloc((size_t)MROWS * DMODEL * 2);
    float*   xs2     = (float*)  alloc((size_t)MROWS * DMODEL * 4);
    __bf16*  h2      = (__bf16*) alloc((size_t)MROWS * DMODEL * 2);
    float*   ffn1    = (float*)  alloc((size_t)MROWS * 4 * DMODEL * 4);
    __bf16*  gbf     = (__bf16*) alloc((size_t)MROWS * 4 * DMODEL * 2);
    float*   xsf     = (float*)  alloc((size_t)MROWS * DMODEL * 4);

    // 1) router + aux
    hipMemsetAsync(auxacc, 0, sizeof(float), stream);
    router_kernel<<<BATCH * TTOK / 8, 256, 0, stream>>>(x, w_router, logits, auxacc);

    // 2) top-k per batch row
    topk_kernel<<<BATCH, 256, 0, stream>>>(logits, idx);

    // 3) weight conversion to bf16
    {
        const int n1 = 3 * DMODEL * DMODEL, n2 = DMODEL * DMODEL, n3 = 4 * DMODEL * DMODEL;
        conv_bf16_kernel<<<(n1 + 255) / 256, 256, 0, stream>>>(in_w,  wqkv, n1);
        conv_bf16_kernel<<<(n2 + 255) / 256, 256, 0, stream>>>(out_w, wout, n2);
        conv_bf16_kernel<<<(n3 + 255) / 256, 256, 0, stream>>>(w1,    w1b,  n3);
        conv_bf16_kernel<<<(n3 + 255) / 256, 256, 0, stream>>>(w2,    w2b,  n3);
    }

    // 4) gather + LN1
    ln_kernel<<<MROWS, 256, 0, stream>>>(x, idx, ln1_g, ln1_b, xs, h1);

    // 5) QKV projection: (8192,1024) x (3072,1024)^T
    gemm_bf16_kernel<<<dim3(3 * DMODEL / 128, MROWS / 128), 256, 0, stream>>>(
        h1, wqkv, in_b, nullptr, qkv, MROWS, 3 * DMODEL, DMODEL);

    // 6) attention
    attn_kernel<<<dim3(KTOK / 64, BATCH * NHEAD), 128, 0, stream>>>(qkv, obf);

    // 7) output projection + residual: xs2 = O*Wo^T + bo + xs
    gemm_bf16_kernel<<<dim3(DMODEL / 128, MROWS / 128), 256, 0, stream>>>(
        obf, wout, out_b, xs, xs2, MROWS, DMODEL, DMODEL);

    // 8) LN2
    ln_kernel<<<MROWS, 256, 0, stream>>>(xs2, nullptr, ln2_g, ln2_b, nullptr, h2);

    // 9) FFN1
    gemm_bf16_kernel<<<dim3(4 * DMODEL / 128, MROWS / 128), 256, 0, stream>>>(
        h2, w1b, b1, nullptr, ffn1, MROWS, 4 * DMODEL, DMODEL);

    // 10) GELU
    {
        const int n = MROWS * 4 * DMODEL;
        gelu_bf16_kernel<<<(n + 255) / 256, 256, 0, stream>>>(ffn1, gbf, n);
    }

    // 11) FFN2 + residual: xsf = g*W2^T + b2 + xs2
    gemm_bf16_kernel<<<dim3(DMODEL / 128, MROWS / 128), 256, 0, stream>>>(
        gbf, w2b, b2, xs2, xsf, MROWS, DMODEL, 4 * DMODEL);

    // 12) out = x ; scatter routed rows ; aux
    hipMemcpyAsync(out, x, (size_t)BATCH * TTOK * DMODEL * sizeof(float),
                   hipMemcpyDeviceToDevice, stream);
    scatter_kernel<<<MROWS, 256, 0, stream>>>(xsf, idx, out);
    aux_kernel<<<1, 1, 0, stream>>>(auxacc, out + (size_t)BATCH * TTOK * DMODEL);
}
